// RingReservoirCell_MinComp_7361573945951
// MI455X (gfx1250) — matrix operand order repaired and checked by reading, not compiled, not run on hardware
//
#include <hip/hip_runtime.h>

typedef __attribute__((ext_vector_type(16))) _Float16 v16h;
typedef __attribute__((ext_vector_type(8)))  _Float16 v8h;
typedef __attribute__((ext_vector_type(8)))  float    v8f;
typedef __attribute__((ext_vector_type(4)))  float    f4;
typedef __attribute__((ext_vector_type(4)))  int      i4;

#define BATCH  32
#define T_LEN  512
#define D_IN   128
#define UNITS  2048
#define TC     32          // timesteps per chunk
#define US     2052        // padded LDS row stride for u (floats); 2052*4 % 16 == 0
#define RHO    0.99f

__device__ __forceinline__ float fast_tanhf(float x) {
#if __has_builtin(__builtin_amdgcn_tanhf)
    return __builtin_amdgcn_tanhf(x);
#elif __has_builtin(__builtin_amdgcn_tanh_f32)
    return __builtin_amdgcn_tanh_f32(x);
#else
    // tanh(x) = 1 - 2/(e^{2x}+1); exact limits at +/-inf, v_exp + v_rcp path
    float e = __expf(2.0f * x);
    return 1.0f - 2.0f * __builtin_amdgcn_rcpf(e + 1.0f);
#endif
}

// One-time: kernel[128][2048] f32  ->  kernelT[2048][128] f16 in d_ws.
__global__ void __launch_bounds__(256) kt_convert(const float* __restrict__ k,
                                                  _Float16* __restrict__ kt) {
    int idx = blockIdx.x * 256 + threadIdx.x;   // 0 .. 2048*128-1
    int d = idx & (D_IN - 1);
    int u = idx >> 7;
    kt[idx] = (_Float16)k[(size_t)d * UNITS + u];
}

// Fused: per-batch workgroup. GEMM chunk (WMMA f16 -> f32) into LDS, then
// sequential ring-reservoir scan over the chunk with NT streaming stores.
__global__ void __launch_bounds__(512, 1) rr_fused(
    const float*    __restrict__ xin,   // [B][T][128]
    const float*    __restrict__ h0,    // [B][2048]
    const _Float16* __restrict__ kt,    // [2048][128] (transposed, f16)
    const float*    __restrict__ bias,  // [2048]
    float*          __restrict__ out)   // [B][T][2048]
{
    __shared__ _Float16 xA[TC * D_IN];     // 8 KB   input chunk, f16
    __shared__ float    uB[TC * US];       // 256.5 KB  u chunk, f32
    __shared__ float    hbuf[2][UNITS];    // 16 KB  double-buffered state

    const int b    = blockIdx.x;
    const int tid  = threadIdx.x;
    const int lane = tid & 31;
    const int wv   = tid >> 5;             // wave 0..15 -> unit slice wv*128..

    // ---- scan-phase ownership: 4 units per thread
    const int j = tid * 4;
    const f4 bj = *(const f4*)(bias + j);

    // ---- init state from h0
    *(f4*)&hbuf[0][j] = *(const f4*)(h0 + (size_t)b * UNITS + j);

    // ---- WMMA lane mapping (A and B share contiguous k order)
    const int mrow   = lane & 15;          // A: M = row index
    const int kbaseA = (lane >> 4) * 16;   // A: lanes 16-31 hold K 16..31
    const int kbaseB = (lane >> 4) * 16;   // B: lanes 16-31 hold K 16..31
    const int ncol   = wv * 128 + (lane & 15);
    const int tb     = (lane >> 4) * 8;    // C/D: vgpr r -> M = tb + r

    const float* xsrc_b = xin + (size_t)b * T_LEN * D_IN;
    float*       out_b  = out + (size_t)b * T_LEN * UNITS;

    int p = 0;
    __syncthreads();

    // Keep the chunk loop rolled: 16 identical iterations separated by
    // barriers; unrolling just multiplies code size 16x (I$ thrash).
    #pragma unroll 1
    for (int t0 = 0; t0 < T_LEN; t0 += TC) {
        // ---------- stage input chunk: f32 global -> f16 LDS ----------
        {
            const float* src = xsrc_b + (size_t)t0 * D_IN;
            const int e = tid * 8;                      // 512 thr * 8 = TC*128
            f4 a = *(const f4*)(src + e);
            f4 c = *(const f4*)(src + e + 4);
            v8h hv = { (_Float16)a.x, (_Float16)a.y, (_Float16)a.z, (_Float16)a.w,
                       (_Float16)c.x, (_Float16)c.y, (_Float16)c.z, (_Float16)c.w };
            *(v8h*)&xA[e] = hv;
            if (t0 + TC < T_LEN)                        // global_prefetch_b8
                __builtin_prefetch(src + TC * D_IN + e, 0, 0);
        }
        __syncthreads();

        // ---------- GEMM chunk: u[TC][2048] = xA(16x32 tiles) * ktT ----------
        // A fragments stay resident (64 VGPRs), reused across all 8 N-tiles.
        v16h afrag[2][4];
        #pragma unroll
        for (int mt = 0; mt < 2; ++mt) {
            const _Float16* rowp = &xA[(mt * 16 + mrow) * D_IN];
            #pragma unroll
            for (int kk = 0; kk < 4; ++kk) {
                union { i4 q[2]; v16h v; } uu;
                uu.q[0] = *(const i4*)(rowp + kk * 32 + kbaseA);        // K kb..kb+7
                uu.q[1] = *(const i4*)(rowp + kk * 32 + kbaseA + 8);    // K kb+8..kb+15
                afrag[mt][kk] = uu.v;
            }
        }
        // Keep this loop rolled: only 4 B fragments (32 VGPRs) live at a time,
        // otherwise clang hoists all 32 fragments and spills to scratch.
        #pragma unroll 1
        for (int nt = 0; nt < 8; ++nt) {
            const int n = ncol + nt * 16;
            const _Float16* bp = kt + (size_t)n * D_IN + kbaseB;
            v16h bf0 = *(const v16h*)(bp);
            v16h bf1 = *(const v16h*)(bp + 32);
            v16h bf2 = *(const v16h*)(bp + 64);
            v16h bf3 = *(const v16h*)(bp + 96);

            v8f acc0 = {0.f,0.f,0.f,0.f,0.f,0.f,0.f,0.f};
            v8f acc1 = {0.f,0.f,0.f,0.f,0.f,0.f,0.f,0.f};
            acc0 = __builtin_amdgcn_wmma_f32_16x16x32_f16(false, afrag[0][0], false, bf0, (short)0, acc0, false, false);
            acc1 = __builtin_amdgcn_wmma_f32_16x16x32_f16(false, afrag[1][0], false, bf0, (short)0, acc1, false, false);
            acc0 = __builtin_amdgcn_wmma_f32_16x16x32_f16(false, afrag[0][1], false, bf1, (short)0, acc0, false, false);
            acc1 = __builtin_amdgcn_wmma_f32_16x16x32_f16(false, afrag[1][1], false, bf1, (short)0, acc1, false, false);
            acc0 = __builtin_amdgcn_wmma_f32_16x16x32_f16(false, afrag[0][2], false, bf2, (short)0, acc0, false, false);
            acc1 = __builtin_amdgcn_wmma_f32_16x16x32_f16(false, afrag[1][2], false, bf2, (short)0, acc1, false, false);
            acc0 = __builtin_amdgcn_wmma_f32_16x16x32_f16(false, afrag[0][3], false, bf3, (short)0, acc0, false, false);
            acc1 = __builtin_amdgcn_wmma_f32_16x16x32_f16(false, afrag[1][3], false, bf3, (short)0, acc1, false, false);

            float* up0 = &uB[tb * US + n];          // m-tile 0: rows tb+r
            float* up1 = &uB[(16 + tb) * US + n];   // m-tile 1: rows 16+tb+r
            #pragma unroll
            for (int r = 0; r < 8; ++r) {
                up0[r * US] = acc0[r];
                up1[r * US] = acc1[r];
            }
        }
        __syncthreads();

        // ---------- sequential ring scan over the chunk ----------
        #pragma unroll 1
        for (int tt = 0; tt < TC; ++tt) {
            const float* hp = hbuf[p];
            float*       hn = hbuf[p ^ 1];
            float hm1 = hp[(j + UNITS - 1) & (UNITS - 1)];
            float ha  = hp[j];
            float hb  = hp[j + 1];
            float hc  = hp[j + 2];
            f4 u4 = *(const f4*)&uB[tt * US + j];
            f4 r;
            r.x = fast_tanhf(u4.x + bj.x + RHO * hm1);
            r.y = fast_tanhf(u4.y + bj.y + RHO * ha);
            r.z = fast_tanhf(u4.z + bj.z + RHO * hb);
            r.w = fast_tanhf(u4.w + bj.w + RHO * hc);
            *(f4*)&hn[j] = r;
            __builtin_nontemporal_store(r, (f4*)(out_b + (size_t)(t0 + tt) * UNITS + j));
            p ^= 1;
            __syncthreads();
        }
    }
}

extern "C" void kernel_launch(void* const* d_in, const int* in_sizes, int n_in,
                              void* d_out, int out_size, void* d_ws, size_t ws_size,
                              hipStream_t stream) {
    (void)in_sizes; (void)n_in; (void)out_size; (void)ws_size;
    const float* xin  = (const float*)d_in[0];   // inputs [32,512,128]
    const float* h0   = (const float*)d_in[1];   // h0     [32,2048]
    const float* kern = (const float*)d_in[2];   // kernel [128,2048]
    const float* bias = (const float*)d_in[3];   // bias   [2048]
    float* out = (float*)d_out;
    _Float16* kt = (_Float16*)d_ws;              // needs 2048*128*2 = 512 KB scratch

    kt_convert<<<(UNITS * D_IN) / 256, 256, 0, stream>>>(kern, kt);
    rr_fused<<<BATCH, 512, 0, stream>>>(xin, h0, kt, bias, out);
}
